// AffineMultiQueryHardAttentionEncoder_41034117546433
// MI455X (gfx1250) — compile-verified, matrix operand order checked by reading, not run on hardware
//
#include <hip/hip_runtime.h>
#include <hip/hip_bf16.h>
#include <math.h>
#include <stdint.h>

// ---------------------------------------------------------------------------
// AffineMultiQueryHardAttentionEncoder for MI455X (gfx1250, wave32)
//   scores[m] = max_n  (q[n]*affine) . keys[m]      (bf16 WMMA, f32 accum)
//   keys streamed HBM -> LDS with global_load_async_to_lds_b128 (ASYNCcnt),
//   double-buffered, overlapped with v_wmma_f32_16x16x32_bf16 compute.
//   sched_group_barrier pipelines each K-step with bounded fragment liveness.
//   top-64 -> softmax -> weighted sum of values rows
// ---------------------------------------------------------------------------

typedef __attribute__((ext_vector_type(16))) __bf16 v16bf;
typedef __attribute__((ext_vector_type(8)))  __bf16 v8bf;
typedef __attribute__((ext_vector_type(8)))  float  v8f;

#define D_DIM 1024
#define N_Q   512
#define KEYS_PER_BLOCK 64          // 4 waves x 16 keys
#define KC    64                   // D-chunk staged in LDS per iteration
#define NKC   (D_DIM / KC)         // 16
#define QG    8                    // query 16-row chunks per accumulator group
#define NGROUPS (N_Q / (16 * QG))  // 4
#define K_TOP 64

#if defined(__has_builtin)
#if __has_builtin(__builtin_amdgcn_sched_group_barrier)
#define HAVE_SCHED_GROUPS 1
#endif
#endif

__device__ __forceinline__ void sched_pipeline_ks() {
#ifdef HAVE_SCHED_GROUPS
  // One K-step, bounded liveness software pipeline:
  //   B-fragment DS reads first, then A-fragment loads staggered so each
  //   WMMA pair executes with the next two load batches in flight.
  __builtin_amdgcn_sched_group_barrier(0x100, 4, 0);  // DS read x4  (B frag)
  __builtin_amdgcn_sched_group_barrier(0x020, 8, 0);  // VMEM read x8 (frags 0-3)
  __builtin_amdgcn_sched_group_barrier(0x008, 2, 0);  // WMMA x2
  __builtin_amdgcn_sched_group_barrier(0x020, 4, 0);  // VMEM read x4 (frags 4-5)
  __builtin_amdgcn_sched_group_barrier(0x008, 2, 0);  // WMMA x2
  __builtin_amdgcn_sched_group_barrier(0x020, 4, 0);  // VMEM read x4 (frags 6-7)
  __builtin_amdgcn_sched_group_barrier(0x008, 2, 0);  // WMMA x2
  __builtin_amdgcn_sched_group_barrier(0x008, 2, 0);  // WMMA x2
#endif
}

__device__ __forceinline__ v16bf load_frag(const __bf16* p0, const __bf16* p1) {
  v16bf r;
  ((v8bf*)&r)[0] = *(const v8bf*)p0;
  ((v8bf*)&r)[1] = *(const v8bf*)p1;
  return r;
}

__device__ __forceinline__ v8f zero8() {
  v8f z = {0.f, 0.f, 0.f, 0.f, 0.f, 0.f, 0.f, 0.f};
  return z;
}

__device__ __forceinline__ v16bf pack16(float4 a, float4 b, float4 c, float4 d) {
  v16bf r;
  r[0]  = (__bf16)a.x; r[1]  = (__bf16)a.y; r[2]  = (__bf16)a.z; r[3]  = (__bf16)a.w;
  r[4]  = (__bf16)b.x; r[5]  = (__bf16)b.y; r[6]  = (__bf16)b.z; r[7]  = (__bf16)b.w;
  r[8]  = (__bf16)c.x; r[9]  = (__bf16)c.y; r[10] = (__bf16)c.z; r[11] = (__bf16)c.w;
  r[12] = (__bf16)d.x; r[13] = (__bf16)d.y; r[14] = (__bf16)d.z; r[15] = (__bf16)d.w;
  return r;
}

// Issue this thread's share (8 x b128) of one 64-key x 64-d fp32 chunk as
// async HBM->LDS copies. Tracked by ASYNCcnt, NOT loadcnt, so the WMMA
// pipeline's A-fragment waits never stall on these.
__device__ __forceinline__ void stage_async(const float* __restrict__ keys,
                                            int M, int mbase, int tid, int kc,
                                            float* dstbase) {
  const uint32_t lds_base = (uint32_t)(uintptr_t)dstbase;  // low 32b = LDS off
#pragma unroll
  for (int i = 0; i < 8; ++i) {
    int linear = i * 128 + tid;          // 0..1023 float4s
    int key    = linear >> 4;            // 16 float4s per key-row-chunk
    int f4     = linear & 15;
    int gm     = mbase + key;
    if (gm >= M) gm = M - 1;             // clamp tail (scores buffer padded)
    uint64_t gaddr =
        (uint64_t)(uintptr_t)(keys + (size_t)gm * D_DIM + kc * KC + f4 * 4);
    uint32_t laddr = lds_base + (uint32_t)((key * KC + f4 * 4) * 4);
    asm volatile("global_load_async_to_lds_b128 %0, %1, off" ::"v"(laddr),
                 "v"(gaddr)
                 : "memory");
  }
}

__device__ __forceinline__ void wait_async0() {
  asm volatile("s_wait_asynccnt 0x0" ::: "memory");
}

// ---- Kernel 1: qa[n][d] = bf16(queries[n][d] * affine[d]) ------------------
__global__ void prep_qa_kernel(const float* __restrict__ q,
                               const float* __restrict__ affine,
                               __bf16* __restrict__ qa, int total) {
  int i = blockIdx.x * blockDim.x + threadIdx.x;
  if (i < total) {
    int d = i & (D_DIM - 1);
    qa[i] = (__bf16)(q[i] * affine[d]);
  }
}

// ---- Kernel 2: scores[m] = max over 512 queries of qa . keys[m] ------------
// Block: 128 threads = 4 waves, 64 keys per block (16 per wave).
// launch_bounds(128, 1): allow full VGPR budget -> no accumulator spills.
__global__ void __launch_bounds__(128, 1)
score_kernel(const float* __restrict__ keys,
             const __bf16* __restrict__ qa,
             float* __restrict__ scores, int M) {
  __shared__ float raw[2][KEYS_PER_BLOCK * KC];   // 2 x 16 KB fp32 key tiles

  const int tid   = threadIdx.x;
  const int lane  = tid & 31;
  const int wave  = tid >> 5;              // 0..3
  const int l16   = lane & 15;
  const int halfo = (lane >> 4) ? 8 : 0;   // 16-bit fragment lane-half offset
  const int mbase = blockIdx.x * KEYS_PER_BLOCK;
  const int nrow  = wave * 16 + l16;       // this lane's key column in block

  float colmax = -INFINITY;

  for (int qg = 0; qg < NGROUPS; ++qg) {
    v8f acc[QG];
#pragma unroll
    for (int i = 0; i < QG; ++i) acc[i] = zero8();

    // Prologue: stage chunk 0 into raw[0].
    stage_async(keys, M, mbase, tid, 0, &raw[0][0]);
    wait_async0();
    __syncthreads();

    for (int kc = 0; kc < NKC; ++kc) {
      const int cur = kc & 1;
      if (kc + 1 < NKC)                      // overlap: next chunk HBM->LDS
        stage_async(keys, M, mbase, tid, kc + 1, &raw[cur ^ 1][0]);

      const float* kbuf = &raw[cur][0];
#pragma unroll
      for (int ks = 0; ks < 2; ++ks) {       // two 16x16x32 K-steps per chunk
        const int r0 = ks * 32 + halfo;
        // B fragment: 16 fp32 from LDS -> bf16 in registers (feeds 8 WMMAs)
        const float* bp = kbuf + nrow * KC + r0;
        float4 b0 = *(const float4*)(bp);
        float4 b1 = *(const float4*)(bp + 4);
        float4 b2 = *(const float4*)(bp + 16);
        float4 b3 = *(const float4*)(bp + 20);
        const v16bf bfrag = pack16(b0, b1, b2, b3);

        // A fragments; sched groups stagger loads ahead of WMMA pairs.
        v16bf afr[QG];
#pragma unroll
        for (int qi = 0; qi < QG; ++qi) {
          const int qrow = (qg * QG + qi) * 16 + l16;
          const __bf16* ap = qa + (size_t)qrow * D_DIM + kc * KC + r0;
          afr[qi] = load_frag(ap, ap + 16);
        }
#pragma unroll
        for (int qi = 0; qi < QG; ++qi) {
          acc[qi] = __builtin_amdgcn_wmma_f32_16x16x32_bf16(
              false, afr[qi], false, bfrag, (short)0, acc[qi], false, false);
        }
        sched_pipeline_ks();
      }

      wait_async0();       // own wave's staging for kc+1 complete
      __syncthreads();     // publish LDS writes; retire reads of raw[cur]
    }

    // Reduce max over the query (M) dimension of each 16x16 tile.
#pragma unroll
    for (int qi = 0; qi < QG; ++qi) {
      float m8 = acc[qi][0];
#pragma unroll
      for (int r = 1; r < 8; ++r) m8 = fmaxf(m8, acc[qi][r]);
      colmax = fmaxf(colmax, m8);
    }
  }

  // C layout: lanes 0-15 hold M=0..7 rows, lanes 16-31 hold M=8..15 of same N.
  colmax = fmaxf(colmax, __shfl_xor(colmax, 16));
  if (lane < 16) {
    scores[mbase + wave * 16 + l16] = colmax;   // scores buffer is padded
  }
}

// ---- Kernel 3: top-64, softmax, weighted sum of value rows -----------------
__global__ void topk_encode_kernel(float* __restrict__ scores,
                                   const float* __restrict__ values,
                                   float* __restrict__ out, int M) {
  __shared__ float rv[1024];
  __shared__ int   ri[1024];
  __shared__ float selw[K_TOP];
  __shared__ int   seli[K_TOP];
  __shared__ float wn[K_TOP];
  const int tid = threadIdx.x;

  for (int it = 0; it < K_TOP; ++it) {
    float best = -INFINITY; int bi = 0x7fffffff;
    for (int m = tid; m < M; m += 1024) {
      float v = scores[m];
      if (v > best) { best = v; bi = m; }
    }
    rv[tid] = best; ri[tid] = bi;
    __syncthreads();
    for (int s = 512; s > 0; s >>= 1) {
      if (tid < s) {
        float v2 = rv[tid + s]; int i2 = ri[tid + s];
        if (v2 > rv[tid] || (v2 == rv[tid] && i2 < ri[tid])) {
          rv[tid] = v2; ri[tid] = i2;
        }
      }
      __syncthreads();
    }
    if (tid == 0) {
      seli[it] = ri[0];
      selw[it] = rv[0];
      scores[ri[0]] = -INFINITY;   // exclude from next round
    }
    __syncthreads();
  }

  // softmax over the 64 selected scores (selw[0] is the max: sorted descending)
  if (tid < K_TOP) wn[tid] = __expf(selw[tid] - selw[0]);
  __syncthreads();
  if (tid == 0) {
    float s = 0.f;
    for (int k = 0; k < K_TOP; ++k) s += wn[k];
    rv[0] = 1.0f / s;
  }
  __syncthreads();
  const float inv = rv[0];

  // encoding[d] = sum_k w[k] * values[idx[k]][d]   (1024 threads, d = tid)
  {
    const int d = tid;
    float e = 0.f;
    for (int k = 0; k < K_TOP; ++k)
      e += wn[k] * inv * values[(size_t)seli[k] * D_DIM + d];
    out[d] = e;
  }
  if (tid < K_TOP) out[D_DIM + tid] = (float)seli[tid];  // exact: M < 2^24
}

// ---------------------------------------------------------------------------
extern "C" void kernel_launch(void* const* d_in, const int* in_sizes, int n_in,
                              void* d_out, int out_size, void* d_ws,
                              size_t ws_size, hipStream_t stream) {
  const float* queries = (const float*)d_in[0];
  const float* keys    = (const float*)d_in[1];
  const float* values  = (const float*)d_in[2];
  const float* affine  = (const float*)d_in[3];
  float* out = (float*)d_out;

  const int M = in_sizes[1] / D_DIM;               // 100000
  const int mblocks = (M + KEYS_PER_BLOCK - 1) / KEYS_PER_BLOCK;

  // workspace layout: [ qa bf16 512x1024 | scores f32 padded ]
  __bf16* qa     = (__bf16*)d_ws;
  float*  scores = (float*)((char*)d_ws + (size_t)N_Q * D_DIM * sizeof(__bf16));

  const int total = N_Q * D_DIM;
  prep_qa_kernel<<<(total + 255) / 256, 256, 0, stream>>>(queries, affine, qa,
                                                          total);
  score_kernel<<<mblocks, 128, 0, stream>>>(keys, qa, scores, M);
  topk_encode_kernel<<<1, 1024, 0, stream>>>(scores, values, out, M);
}
